// MoEClassifier_86079734546616
// MI455X (gfx1250) — matrix operand order repaired and checked by reading, not compiled, not run on hardware
//
#include <hip/hip_runtime.h>
#include <hip/hip_bf16.h>
#include <math.h>
#include <stdint.h>

// Problem sizes (fixed by reference)
#define NTOK 16384
#define DIN  1024
#define DH   1024
#define NCLS 512
#define NEXP 8

typedef __attribute__((ext_vector_type(16))) _Float16 v16h;
typedef __attribute__((ext_vector_type(8)))  _Float16 v8h;
typedef __attribute__((ext_vector_type(8)))  float    v8f;

#if defined(__has_builtin)
#  if __has_builtin(__builtin_amdgcn_tensor_load_to_lds)
#    define HAVE_TDM 1
#  endif
#endif
#ifndef HAVE_TDM
#  define HAVE_TDM 0
#endif

static __device__ inline v16h concat16(v8h lo, v8h hi) {
    return __builtin_shufflevector(lo, hi, 0,1,2,3,4,5,6,7,8,9,10,11,12,13,14,15);
}

static __device__ inline float gelu_erf(float x) {
    return 0.5f * x * (1.0f + erff(x * 0.70710678118654752440f));
}

// Per-lane async global->LDS 16B copy (no VGPR round-trip, tracked by ASYNCcnt)
static __device__ inline void async_copy16(unsigned lds_off, const void* gaddr) {
    asm volatile("global_load_async_to_lds_b128 %0, %1, off"
                 :: "v"(lds_off), "v"(gaddr) : "memory");
}
static __device__ inline void async_wait0() {
    asm volatile("s_wait_asynccnt 0" ::: "memory");
}

#if HAVE_TDM
typedef unsigned int v4u  __attribute__((ext_vector_type(4)));
typedef int          v8i_t __attribute__((ext_vector_type(8)));
typedef int          v4i_t __attribute__((ext_vector_type(4)));

// TDM 2D tile load: tile (tile1 rows x tile0 elems) of 2-byte data from a
// row-major tensor (row stride stride0 elems) into LDS with 4-DWORD padding
// after every 16 DWORDs (i.e. +8 elems pad per 32-elem row -> 40-elem stride).
static __device__ inline void tdm_load_2d(unsigned lds_addr, const void* gaddr,
                                          unsigned dim0, unsigned dim1,
                                          unsigned tile0, unsigned tile1,
                                          unsigned stride0) {
    unsigned long long ga = (unsigned long long)(uintptr_t)gaddr;
    v4u g0;
    g0.x = 1u;                                   // count=1, user descriptor
    g0.y = lds_addr;                             // LDS byte address
    g0.z = (unsigned)ga;                         // global addr lo32
    g0.w = (unsigned)(ga >> 32) | (2u << 30);    // addr hi25 | type=2 (image)
    v8i_t g1;
    g1[0] = (int)((1u << 16)      // data_size = 1 -> 2 bytes
                | (1u << 20)      // pad_enable
                | (3u << 22)      // pad_interval code 3 -> 16 DWORDs
                | (3u << 25));    // pad_amount  code 3 -> 4 DWORDs
    g1[1] = (int)(dim0 << 16);                       // tensor_dim0 lo16
    g1[2] = (int)((dim0 >> 16) | (dim1 << 16));      // dim0 hi16 | dim1 lo16
    g1[3] = (int)((dim1 >> 16) | (tile0 << 16));     // dim1 hi16 | tile_dim0
    g1[4] = (int)(tile1 & 0xFFFFu);                  // tile_dim1 | tile_dim2=0
    g1[5] = (int)stride0;                            // tensor_dim0_stride lo32
    g1[6] = 0;                                       // stride0 hi16 | dim1_stride
    g1[7] = 0;
    v4i_t z4 = {0, 0, 0, 0};
    v8i_t z8 = {0, 0, 0, 0, 0, 0, 0, 0};
    // clang-23 / therock-10.0 form: 6 args (g0, g1, g2, g3, ext, cpol)
    __builtin_amdgcn_tensor_load_to_lds(g0, g1, z4, z4, z8, 0);
}
#endif

// ---------------------------------------------------------------------------
// Conversion kernels: fp32 -> fp16 (plain, and transpose [K][N] -> [N][K])
// ---------------------------------------------------------------------------
__global__ void cvt_f32_f16(const float* __restrict__ src,
                            _Float16* __restrict__ dst, long n) {
    long i = (long)blockIdx.x * blockDim.x + threadIdx.x;
    long stride = (long)gridDim.x * blockDim.x;
    for (; i < n; i += stride) dst[i] = (_Float16)src[i];
}

// src: [batch][K][N] fp32 (row-major) -> dst: [batch][N][K] fp16
__global__ void cvt_transpose_f16(const float* __restrict__ src,
                                  _Float16* __restrict__ dst, int K, int N) {
    long base = (long)blockIdx.y * (long)K * (long)N;
    long tot  = (long)K * (long)N;
    long i = (long)blockIdx.x * blockDim.x + threadIdx.x;
    long stride = (long)gridDim.x * blockDim.x;
    for (; i < tot; i += stride) {
        int n = (int)(i / K);
        int k = (int)(i % K);
        dst[base + i] = (_Float16)src[base + (long)k * N + n];
    }
}

// ---------------------------------------------------------------------------
// Kernel 1: h = gelu(x @ proj_w + proj_b), fp16 in, f32 accum, fp16 out
// Block tile 128x128, TK=32. 8 waves: 4 in M x 2 in N, wave tile 32x64.
// Staging via global_load_async_to_lds_b128 (ASYNCcnt).
// ---------------------------------------------------------------------------
#define PBM 128
#define PBN 128
#define PTK 32
#define LPAD 8
#define LSTR (PTK + LPAD)   // 40 elems = 80B row stride (16B aligned, conflict-free)

__global__ void __launch_bounds__(256)
proj_gemm_gelu(const _Float16* __restrict__ X,   // [NTOK][DIN]
               const _Float16* __restrict__ Wt,  // [DH][DIN]  (transposed)
               const float*    __restrict__ bias,// [DH]
               _Float16*       __restrict__ H)   // [NTOK][DH]
{
    __shared__ _Float16 lA[PBM * LSTR];
    __shared__ _Float16 lB[PBN * LSTR];

    const int tid    = threadIdx.x;
    const int lane   = tid & 31;
    const int wave   = tid >> 5;
    const int lr     = lane & 15;
    const int laneHi = lane >> 4;        // 0: lanes 0-15, 1: lanes 16-31
    const int waveM  = wave & 3;         // 0..3
    const int waveN  = wave >> 2;        // 0..1

    const int m0 = blockIdx.x * PBM;
    const int n0 = blockIdx.y * PBN;

    v8f acc[2][4];
    const v8f vzero = {0.f,0.f,0.f,0.f,0.f,0.f,0.f,0.f};
    #pragma unroll
    for (int i = 0; i < 2; ++i)
        #pragma unroll
        for (int j = 0; j < 4; ++j) acc[i][j] = vzero;

    for (int kt = 0; kt < DIN; kt += PTK) {
        // Stage A (x tile) and B (W^T tile): 512 16B async copies each
        #pragma unroll
        for (int c = 0; c < 2; ++c) {
            int ch = tid + c * 256;
            int r  = ch >> 2;            // 0..127
            int kc = (ch & 3) * 8;       // 0,8,16,24
            async_copy16((unsigned)(uintptr_t)&lA[r * LSTR + kc],
                         &X[(long)(m0 + r) * DIN + kt + kc]);
            async_copy16((unsigned)(uintptr_t)&lB[r * LSTR + kc],
                         &Wt[(long)(n0 + r) * DIN + kt + kc]);
            // speculative prefetch of next K tile (global_prefetch_b8)
            __builtin_prefetch(&X[(long)(m0 + r) * DIN + kt + PTK + kc]);
        }
        async_wait0();
        __syncthreads();

        // Build fragments per ISA 7.12.2 layouts
        v16h a[2], b[4];
        #pragma unroll
        for (int i = 0; i < 2; ++i) {
            int am = waveM * 32 + i * 16 + lr;
            const _Float16* p = &lA[am * LSTR + laneHi * 8];
            a[i] = concat16(*(const v8h*)p, *(const v8h*)(p + 16));
        }
        #pragma unroll
        for (int j = 0; j < 4; ++j) {
            int bn = waveN * 64 + j * 16 + lr;
            const _Float16* p = &lB[bn * LSTR + laneHi * 8];
            b[j] = concat16(*(const v8h*)p, *(const v8h*)(p + 16));
        }
        #pragma unroll
        for (int i = 0; i < 2; ++i)
            #pragma unroll
            for (int j = 0; j < 4; ++j)
                acc[i][j] = __builtin_amdgcn_wmma_f32_16x16x32_f16(
                    false, a[i], false, b[j], (short)0, acc[i][j], false, false);
        __syncthreads();
    }

    // Epilogue: bias + exact GELU, store fp16 h
    #pragma unroll
    for (int i = 0; i < 2; ++i) {
        int mb = m0 + waveM * 32 + i * 16 + laneHi * 8;
        #pragma unroll
        for (int j = 0; j < 4; ++j) {
            int n  = n0 + waveN * 64 + j * 16 + lr;
            float bn = bias[n];
            #pragma unroll
            for (int r = 0; r < 8; ++r) {
                float v = acc[i][j][r] + bn;
                H[(long)(mb + r) * DH + n] = (_Float16)gelu_erf(v);
            }
        }
    }
}

// ---------------------------------------------------------------------------
// Kernel 2: gating. One block per token: logits, softmax, top-2 renorm.
// ---------------------------------------------------------------------------
__global__ void __launch_bounds__(256)
gate_kernel(const _Float16* __restrict__ H,
            const float* __restrict__ gw,   // [DH][NEXP]
            const float* __restrict__ gb,   // [NEXP]
            float* __restrict__ dense,      // [NTOK][NEXP]
            float* __restrict__ probs)      // [NTOK][NEXP]
{
    __shared__ float red[NEXP][256];
    const int t   = blockIdx.x;
    const int tid = threadIdx.x;

    float acc[NEXP];
    #pragma unroll
    for (int e = 0; e < NEXP; ++e) acc[e] = 0.f;

    for (int d = tid; d < DH; d += 256) {
        float hv = (float)H[(long)t * DH + d];
        #pragma unroll
        for (int e = 0; e < NEXP; ++e) acc[e] += hv * gw[d * NEXP + e];
    }
    #pragma unroll
    for (int e = 0; e < NEXP; ++e) red[e][tid] = acc[e];
    __syncthreads();
    for (int s = 128; s > 0; s >>= 1) {
        if (tid < s) {
            #pragma unroll
            for (int e = 0; e < NEXP; ++e) red[e][tid] += red[e][tid + s];
        }
        __syncthreads();
    }
    if (tid == 0) {
        float lg[NEXP];
        #pragma unroll
        for (int e = 0; e < NEXP; ++e) lg[e] = red[e][0] + gb[e];
        float mx = lg[0];
        #pragma unroll
        for (int e = 1; e < NEXP; ++e) mx = fmaxf(mx, lg[e]);
        float ex[NEXP], sum = 0.f;
        #pragma unroll
        for (int e = 0; e < NEXP; ++e) { ex[e] = expf(lg[e] - mx); sum += ex[e]; }
        float p[NEXP];
        #pragma unroll
        for (int e = 0; e < NEXP; ++e) p[e] = ex[e] / sum;
        int i1 = 0;
        #pragma unroll
        for (int e = 1; e < NEXP; ++e) if (p[e] > p[i1]) i1 = e;
        int i2 = (i1 == 0) ? 1 : 0;
        #pragma unroll
        for (int e = 0; e < NEXP; ++e) if (e != i1 && p[e] > p[i2]) i2 = e;
        float t12 = p[i1] + p[i2];
        #pragma unroll
        for (int e = 0; e < NEXP; ++e) {
            probs[(long)t * NEXP + e] = p[e];
            dense[(long)t * NEXP + e] = 0.f;
        }
        dense[(long)t * NEXP + i1] = p[i1] / t12;
        dense[(long)t * NEXP + i2] = p[i2] / t12;
    }
}

// ---------------------------------------------------------------------------
// Kernel 3: out = sum_e dense[:,e] * (h @ expert_w[e] + expert_b[e])
// Block tile 64x64; H tile + all 8 experts' W tiles staged by the Tensor
// Data Mover (TENSORcnt) with LDS padding matching LSTR. Experts inner loop.
// ---------------------------------------------------------------------------
#define EBM 64
#define EBN 64

__global__ void __launch_bounds__(256)
expert_gemm(const _Float16* __restrict__ H,   // [NTOK][DH]
            const _Float16* __restrict__ Wt,  // [NEXP][NCLS][DH] (transposed)
            const float* __restrict__ eb,     // [NEXP][NCLS]
            const float* __restrict__ dw,     // [NTOK][NEXP]
            float* __restrict__ out)          // [NTOK][NCLS]
{
    __shared__ _Float16 lH[EBM * LSTR];          //  5 KB
    __shared__ _Float16 lW[NEXP * EBN * LSTR];   // 40 KB

    const int tid    = threadIdx.x;
    const int lane   = tid & 31;
    const int wave   = tid >> 5;
    const int lr     = lane & 15;
    const int laneHi = lane >> 4;
    const int waveM  = wave & 1;     // 0..1
    const int waveN  = wave >> 1;    // 0..3

    const int m0 = blockIdx.x * EBM;
    const int n0 = blockIdx.y * EBN;

    v8f acc[NEXP][2];
    const v8f vzero = {0.f,0.f,0.f,0.f,0.f,0.f,0.f,0.f};
    #pragma unroll
    for (int e = 0; e < NEXP; ++e) { acc[e][0] = vzero; acc[e][1] = vzero; }

    for (int kt = 0; kt < DH; kt += PTK) {
#if HAVE_TDM
        // TDM: 9 descriptors per K-step (1 H tile + 8 expert W tiles),
        // issued by wave 0, completion via TENSORcnt.
        if (wave == 0) {
            tdm_load_2d((unsigned)(uintptr_t)&lH[0],
                        &H[(long)m0 * DH + kt],
                        DH, NTOK, PTK, EBM, DH);
            #pragma unroll
            for (int e = 0; e < NEXP; ++e)
                tdm_load_2d((unsigned)(uintptr_t)&lW[e * EBN * LSTR],
                            &Wt[((long)e * NCLS + n0) * DH + kt],
                            DH, NCLS, PTK, EBN, DH);
            __builtin_amdgcn_s_wait_tensorcnt(0);
        }
#else
        // Fallback: per-lane async copies (ASYNCcnt)
        {
            int r  = tid >> 2;
            int kc = (tid & 3) * 8;
            async_copy16((unsigned)(uintptr_t)&lH[r * LSTR + kc],
                         &H[(long)(m0 + r) * DH + kt + kc]);
        }
        #pragma unroll
        for (int c = 0; c < 8; ++c) {
            int ch = tid + c * 256;
            int e  = ch >> 8;
            int r  = (ch & 255) >> 2;
            int kc = (ch & 3) * 8;
            async_copy16((unsigned)(uintptr_t)&lW[(e * EBN + r) * LSTR + kc],
                         &Wt[((long)e * NCLS + n0 + r) * DH + kt + kc]);
        }
        async_wait0();
#endif
        __syncthreads();

        v16h a[2];
        #pragma unroll
        for (int i = 0; i < 2; ++i) {
            int am = waveM * 32 + i * 16 + lr;
            const _Float16* p = &lH[am * LSTR + laneHi * 8];
            a[i] = concat16(*(const v8h*)p, *(const v8h*)(p + 16));
        }
        int bn = waveN * 16 + lr;
        #pragma unroll
        for (int e = 0; e < NEXP; ++e) {
            const _Float16* p = &lW[(e * EBN + bn) * LSTR + laneHi * 8];
            v16h b = concat16(*(const v8h*)p, *(const v8h*)(p + 16));
            #pragma unroll
            for (int i = 0; i < 2; ++i)
                acc[e][i] = __builtin_amdgcn_wmma_f32_16x16x32_f16(
                    false, a[i], false, b, (short)0, acc[e][i], false, false);
        }
        __syncthreads();
    }

    // Epilogue: out[m][n] = sum_e dw[m][e] * (acc_e[m][n] + eb[e][n])
    const int n = n0 + waveN * 16 + lr;
    #pragma unroll
    for (int i = 0; i < 2; ++i) {
        int mb = m0 + waveM * 32 + i * 16 + laneHi * 8;
        float fin[8];
        #pragma unroll
        for (int r = 0; r < 8; ++r) fin[r] = 0.f;
        #pragma unroll
        for (int e = 0; e < NEXP; ++e) {
            float be = eb[e * NCLS + n];
            #pragma unroll
            for (int r = 0; r < 8; ++r) {
                float s = dw[(long)(mb + r) * NEXP + e];
                fin[r] += s * (acc[e][i][r] + be);
            }
        }
        #pragma unroll
        for (int r = 0; r < 8; ++r)
            out[(long)(mb + r) * NCLS + n] = fin[r];
    }
}

// ---------------------------------------------------------------------------
// Kernel 4: deterministic aux loss (fixed-order reductions, no fp atomics)
// ---------------------------------------------------------------------------
__global__ void aux_kernel(const float* __restrict__ dense,
                           const float* __restrict__ probs,
                           float* __restrict__ out_aux) {
    __shared__ float fr[NEXP], mp[NEXP];
    int e = threadIdx.x;
    if (e < NEXP) {
        float f = 0.f, m = 0.f;
        for (int t = 0; t < NTOK; ++t) {
            f += (dense[(long)t * NEXP + e] > 0.f) ? 1.f : 0.f;
            m += probs[(long)t * NEXP + e];
        }
        fr[e] = f / (float)NTOK;
        mp[e] = m / (float)NTOK;
    }
    __syncthreads();
    if (e == 0) {
        float s = 0.f;
        for (int i = 0; i < NEXP; ++i) s += fr[i] * mp[i];
        *out_aux = (float)NEXP * s;
    }
}

// ---------------------------------------------------------------------------
// Launch
// ---------------------------------------------------------------------------
extern "C" void kernel_launch(void* const* d_in, const int* in_sizes, int n_in,
                              void* d_out, int out_size, void* d_ws, size_t ws_size,
                              hipStream_t stream) {
    const float* x        = (const float*)d_in[0];   // [NTOK][DIN]
    const float* proj_w   = (const float*)d_in[1];   // [DIN][DH]
    const float* proj_b   = (const float*)d_in[2];   // [DH]
    const float* gate_w   = (const float*)d_in[3];   // [DH][NEXP]
    const float* gate_b   = (const float*)d_in[4];   // [NEXP]
    const float* expert_w = (const float*)d_in[5];   // [NEXP][DH][NCLS]
    const float* expert_b = (const float*)d_in[6];   // [NEXP][NCLS]
    float* out = (float*)d_out;

    // Workspace layout (fp16 staging + gating buffers), ~75 MB total
    char* ws = (char*)d_ws;
    _Float16* xh  = (_Float16*)(ws);                            // 32 MB
    _Float16* pwt = (_Float16*)(ws + 33554432);                 //  2 MB
    _Float16* ewt = (_Float16*)(ws + 35651584);                 //  8 MB
    _Float16* hh  = (_Float16*)(ws + 44040192);                 // 32 MB
    float*    dns = (float*)   (ws + 77594624);                 // 512 KB
    float*    prb = (float*)   (ws + 78118912);                 // 512 KB

    cvt_f32_f16<<<2048, 256, 0, stream>>>(x, xh, (long)NTOK * DIN);
    cvt_transpose_f16<<<dim3(1024, 1), 256, 0, stream>>>(proj_w, pwt, DIN, DH);
    cvt_transpose_f16<<<dim3(1024, NEXP), 256, 0, stream>>>(expert_w, ewt, DH, NCLS);

    proj_gemm_gelu<<<dim3(NTOK / PBM, DH / PBN), 256, 0, stream>>>(xh, pwt, proj_b, hh);
    gate_kernel<<<NTOK, 256, 0, stream>>>(hh, gate_w, gate_b, dns, prb);
    expert_gemm<<<dim3(NTOK / EBM, NCLS / EBN), 256, 0, stream>>>(hh, ewt, expert_b, dns, out);
    aux_kernel<<<1, 64, 0, stream>>>(dns, prb, out + (long)NTOK * NCLS);
}
